// qkv_attn_11519102288420
// MI455X (gfx1250) — matrix operand order repaired and checked by reading, not compiled, not run on hardware
//
#include <hip/hip_runtime.h>
#include <hip/hip_bf16.h>
#include <math.h>

// ---------------- problem constants ----------------
#define MIDC   256
#define NHEADS 16
#define HEADC  16
#define HWDIM  16384
#define NBATCH 8
#define NROWS  (NBATCH * HWDIM)   // 131072
#define MBLK   128                 // rows per workgroup
#define LSTR   264                 // LDS row stride in bf16 halfs (16B-aligned, conflict-padded)
#define INV_HW (1.0f / 16384.0f)
#define LN_EPS 1e-5f

// ---------------- workspace layout (bytes) ----------------
#define WS_WKV    0            // packed bf16 W_kv fragments: 512*256*2   = 262144
#define WS_WPROJ  262144       // packed bf16 W_proj fragments: 256*256*2 = 131072
#define WS_KVACC  393216       // fp32 kv accumulator: 8*16*16*16*4       = 131072
#define WS_WEFF   524288       // packed bf16 Weff[b] fragments: 8*65536*2= 1048576
#define WS_BEFF   1572864      // fp32 beff: 8*256*4                      = 8192
// total ~1.59 MB

typedef __attribute__((ext_vector_type(16))) __bf16 v16bf;
typedef __attribute__((ext_vector_type(8)))  float  v8f;

// ---------------- helpers ----------------
__device__ __forceinline__ unsigned pack_bf16x2(float a, float b) {
    union { __bf16 h[2]; unsigned u; } t;
    t.h[0] = (__bf16)a; t.h[1] = (__bf16)b;
    return t.u;
}

// A fragment (16x32 bf16, M x K) from LDS, row-major with LSTR stride.
// Layout: lane<16 -> M=lane, K in {0..15}; lane>=16 -> M=lane-16, K in {16..31}
__device__ __forceinline__ v16bf load_afrag_lds(const __bf16* lds, int row_base, int lane, int k0) {
    const int m  = lane & 15;
    const int kb = (lane >> 4) * 16;
    const __bf16* rp = lds + (size_t)(row_base + m) * LSTR + k0 + kb;
    union { v16bf v; uint4 q[2]; } f;
    f.q[0] = *(const uint4*)(rp);
    f.q[1] = *(const uint4*)(rp + 8);
    return f.v;
}

// B fragment (32x16 bf16, K x N) from a pre-packed global buffer:
// per tile, lane holds its 16 halfs contiguously (32 bytes).
__device__ __forceinline__ v16bf load_bfrag(const __bf16* pack, int tile_linear, int lane) {
    const __bf16* p = pack + ((size_t)tile_linear * 32 + lane) * 16;
    union { v16bf v; uint4 q[2]; } f;
    f.q[0] = ((const uint4*)p)[0];
    f.q[1] = ((const uint4*)p)[1];
    return f.v;
}

__device__ __forceinline__ float hsum16(float v) {
    v += __shfl_xor(v, 1);
    v += __shfl_xor(v, 2);
    v += __shfl_xor(v, 4);
    v += __shfl_xor(v, 8);
    return v;
}

// stage 128 x 256 fp32 rows -> LDS bf16 [128][LSTR]
__device__ __forceinline__ void stage_x_bf16(const float* __restrict__ x, int row0,
                                             __bf16* __restrict__ lds, int tid) {
    const int r  = tid & 127;
    const int ch = (tid >> 7) * 128;
    const float4* xv = (const float4*)(x + (size_t)(row0 + r) * MIDC + ch);
    uint2* dst = (uint2*)(lds + (size_t)r * LSTR + ch);
#pragma unroll
    for (int i = 0; i < 32; ++i) {
        float4 f = xv[i];
        uint2 p;
        p.x = pack_bf16x2(f.x, f.y);
        p.y = pack_bf16x2(f.z, f.w);
        dst[i] = p;
    }
}

// ---------------- kernel: pack W_kv and W_proj into bf16 B-fragments ----------------
// B[k][n] = W[o(n)][k].  Fragment order: lane l of tile (nt,kt): n = nt*16+(l&15),
// K covered = kt*32 + ((l>>4)*16) + j, j=0..15, stored contiguously.
__global__ void pack_weights_kernel(const float* __restrict__ Wqkv,
                                    const float* __restrict__ Wproj,
                                    __bf16* __restrict__ wkv_pack,
                                    __bf16* __restrict__ wproj_pack) {
    const int gid  = blockIdx.x * blockDim.x + threadIdx.x;
    const int tile = gid >> 5;
    const int lane = gid & 31;
    const int kb   = (lane >> 4) * 16;
    union { __bf16 h[16]; uint4 q[2]; } o;
    if (tile < 256) {                 // W_kv: NT=32 (512 cols), KT=8
        const int nt = tile >> 3, kt = tile & 7;
        const int np = nt * 16 + (lane & 15);        // 0..511
        const int h  = np >> 5, t = np & 31;
        const int orow = 48 * h + 16 + t;            // skip the q slice
#pragma unroll
        for (int j = 0; j < 16; ++j)
            o.h[j] = (__bf16)Wqkv[(size_t)orow * MIDC + kt * 32 + kb + j];
        __bf16* dst = wkv_pack + ((size_t)tile * 32 + lane) * 16;
        ((uint4*)dst)[0] = o.q[0]; ((uint4*)dst)[1] = o.q[1];
    } else if (tile < 384) {          // W_proj: NT=16, KT=8
        const int tt = tile - 256;
        const int nt = tt >> 3, kt = tt & 7;
        const int orow = nt * 16 + (lane & 15);
#pragma unroll
        for (int j = 0; j < 16; ++j)
            o.h[j] = (__bf16)Wproj[(size_t)orow * MIDC + kt * 32 + kb + j];
        __bf16* dst = wproj_pack + ((size_t)tt * 32 + lane) * 16;
        ((uint4*)dst)[0] = o.q[0]; ((uint4*)dst)[1] = o.q[1];
    }
}

// ---------------- kernel 1: k/v GEMM + LayerNorm + partial kv accumulation ----------------
__global__ void qkv_ln_kv_kernel(const float* __restrict__ x,
                                 const __bf16* __restrict__ wkv_pack,
                                 const float* __restrict__ b_qkv,
                                 const float* __restrict__ kln_w, const float* __restrict__ kln_b,
                                 const float* __restrict__ vln_w, const float* __restrict__ vln_b,
                                 float* __restrict__ kv_acc) {
    extern __shared__ char smem[];
    __bf16* xlds  = (__bf16*)smem;                         // 128*264*2 = 67584
    float*  kvlds = (float*)(smem + 67584);                // 16*16*16*4 = 16384
    float*  stage = (float*)(smem + 67584 + 16384);        // 8 waves * 544 floats

    const int tid  = threadIdx.x;
    const int w    = tid >> 5;
    const int lane = tid & 31;
    const int row0 = blockIdx.x * MBLK;
    const int bat  = row0 >> 14;   // /HWDIM

    __builtin_prefetch(wkv_pack, 0, 3);

    for (int i = tid; i < NHEADS * HEADC * HEADC; i += 256) kvlds[i] = 0.0f;
    stage_x_bf16(x, row0, xlds, tid);
    __syncthreads();

    float* kbuf = stage + w * 544;     // [16][17]
    float* vbuf = kbuf + 272;          // [16][17]

    const int col = lane & 15;
    const int hi8 = (lane >> 4) << 3;

    for (int h = 0; h < NHEADS; ++h) {
        v8f acck = {0.f,0.f,0.f,0.f,0.f,0.f,0.f,0.f};
        v8f accv = {0.f,0.f,0.f,0.f,0.f,0.f,0.f,0.f};
#pragma unroll
        for (int kk = 0; kk < 8; ++kk) {
            v16bf a  = load_afrag_lds(xlds, 16 * w, lane, kk * 32);
            v16bf bk = load_bfrag(wkv_pack, (2 * h) * 8 + kk, lane);
            v16bf bv = load_bfrag(wkv_pack, (2 * h + 1) * 8 + kk, lane);
            acck = __builtin_amdgcn_wmma_f32_16x16x32_bf16(false, a, false, bk, (short)0, acck, false, false);
            accv = __builtin_amdgcn_wmma_f32_16x16x32_bf16(false, a, false, bv, (short)0, accv, false, false);
        }
        // LayerNorm (unbiased std, div by std+eps) per row of 16, then stage to LDS [n][d]
        const float kw  = kln_w[h * HEADC + col], kb2 = kln_b[h * HEADC + col];
        const float vw  = vln_w[h * HEADC + col], vb2 = vln_b[h * HEADC + col];
        const float bk0 = b_qkv[48 * h + 16 + col];
        const float bv0 = b_qkv[48 * h + 32 + col];
#pragma unroll
        for (int r = 0; r < 8; ++r) {
            const int n = r + hi8;
            {
                float val = acck[r] + bk0;
                float s1 = hsum16(val), s2 = hsum16(val * val);
                float mean = s1 * (1.f / 16.f);
                float var  = (s2 - s1 * mean) * (1.f / 15.f);
                float stdv = sqrtf(fmaxf(var, 0.f));
                kbuf[n * 17 + col] = ((val - mean) / (stdv + LN_EPS)) * kw + kb2;
            }
            {
                float val = accv[r] + bv0;
                float s1 = hsum16(val), s2 = hsum16(val * val);
                float mean = s1 * (1.f / 16.f);
                float var  = (s2 - s1 * mean) * (1.f / 15.f);
                float stdv = sqrtf(fmaxf(var, 0.f));
                vbuf[n * 17 + col] = ((val - mean) / (stdv + LN_EPS)) * vw + vb2;
            }
        }
        asm volatile("" ::: "memory");   // wave-internal LDS ordering fence (HW DS is in-order per wave)
        // partial kv[d][e] += sum_n k[n][d]*v[n][e] over this wave's 16 rows
        const int e  = lane & 15;
        const int db = hi8;
#pragma unroll
        for (int j = 0; j < 8; ++j) {
            const int d = db + j;
            float s = 0.f;
#pragma unroll
            for (int n = 0; n < 16; ++n) s += kbuf[n * 17 + d] * vbuf[n * 17 + e];
            atomicAdd(&kvlds[h * 256 + d * 16 + e], s);   // ds_add_f32
        }
        asm volatile("" ::: "memory");
    }
    __syncthreads();
    // flush block-partial kv to global accumulator
    for (int i = tid; i < NHEADS * HEADC * HEADC; i += 256)
        atomicAdd(&kv_acc[(size_t)bat * 4096 + i], kvlds[i]);
}

// ---------------- kernel: Weff_b[c][16h+e] = sum_d Wq[48h+d][c] * kv[b][h][d][e]/HW ----------------
__global__ void weff_kernel(const float* __restrict__ Wqkv,
                            const float* __restrict__ kv_acc,
                            __bf16* __restrict__ weff_pack) {
    const int gid  = blockIdx.x * blockDim.x + threadIdx.x;  // 8 * 128 tiles * 32 lanes
    const int bat  = gid >> 12;
    const int rem  = gid & 4095;
    const int tile = rem >> 5;
    const int lane = rem & 31;
    const int nt = tile >> 3, kt = tile & 7;
    const int np = nt * 16 + (lane & 15);
    const int h  = np >> 4, e = np & 15;
    const int kb = (lane >> 4) * 16;
    float kvv[16];
#pragma unroll
    for (int d = 0; d < 16; ++d)
        kvv[d] = kv_acc[(size_t)bat * 4096 + h * 256 + d * 16 + e] * INV_HW;
    union { __bf16 hh[16]; uint4 q[2]; } o;
#pragma unroll
    for (int j = 0; j < 16; ++j) {
        const int c = kt * 32 + kb + j;
        float s = 0.f;
#pragma unroll
        for (int d = 0; d < 16; ++d) s += Wqkv[(size_t)(48 * h + d) * MIDC + c] * kvv[d];
        o.hh[j] = (__bf16)s;
    }
    __bf16* dst = weff_pack + (size_t)bat * 65536 + ((size_t)tile * 32 + lane) * 16;
    ((uint4*)dst)[0] = o.q[0]; ((uint4*)dst)[1] = o.q[1];
}

__global__ void beff_kernel(const float* __restrict__ b_qkv,
                            const float* __restrict__ kv_acc,
                            float* __restrict__ beff) {
    const int bat = blockIdx.x;
    const int ep  = threadIdx.x;         // 0..255
    const int h = ep >> 4, e = ep & 15;
    float s = 0.f;
#pragma unroll
    for (int d = 0; d < 16; ++d)
        s += b_qkv[48 * h + d] * kv_acc[(size_t)bat * 4096 + h * 256 + d * 16 + e];
    beff[bat * MIDC + ep] = s * INV_HW;
}

// ---------------- kernel 3: ret = x + x@Weff_b + beff ; y = gelu(ret@Wproj^T + bp) + x ----------------
__global__ void fused_out_kernel(const float* __restrict__ x,
                                 const __bf16* __restrict__ weff_pack,
                                 const float* __restrict__ beff,
                                 const __bf16* __restrict__ wproj_pack,
                                 const float* __restrict__ b_proj,
                                 float* __restrict__ out) {
    extern __shared__ char smem[];
    __bf16* xlds   = (__bf16*)smem;                  // 67584
    __bf16* retlds = (__bf16*)(smem + 67584);        // 67584

    const int tid  = threadIdx.x;
    const int w    = tid >> 5;
    const int lane = tid & 31;
    const int row0 = blockIdx.x * MBLK;
    const int bat  = row0 >> 14;
    const int col0 = lane & 15;
    const int hi8  = (lane >> 4) << 3;

    const __bf16* weffb = weff_pack + (size_t)bat * 65536;
    __builtin_prefetch(weffb, 0, 3);
    __builtin_prefetch(wproj_pack, 0, 3);

    stage_x_bf16(x, row0, xlds, tid);
    __syncthreads();

    // Phase A: ret block into LDS (bf16)
#pragma unroll 1
    for (int pass = 0; pass < 2; ++pass) {
        v8f acc[8];
#pragma unroll
        for (int t = 0; t < 8; ++t) acc[t] = (v8f){0.f,0.f,0.f,0.f,0.f,0.f,0.f,0.f};
#pragma unroll
        for (int kk = 0; kk < 8; ++kk) {
            v16bf a = load_afrag_lds(xlds, 16 * w, lane, kk * 32);
#pragma unroll
            for (int t = 0; t < 8; ++t) {
                v16bf bb = load_bfrag(weffb, (pass * 8 + t) * 8 + kk, lane);
                acc[t] = __builtin_amdgcn_wmma_f32_16x16x32_bf16(false, a, false, bb, (short)0, acc[t], false, false);
            }
        }
#pragma unroll
        for (int t = 0; t < 8; ++t) {
            const int col = (pass * 8 + t) * 16 + col0;
            const float be = beff[bat * MIDC + col];
#pragma unroll
            for (int r = 0; r < 8; ++r) {
                const int lr = 16 * w + r + hi8;
                const size_t row = (size_t)row0 + lr;
                const float rv = acc[t][r] + be + x[row * MIDC + col];
                retlds[(size_t)lr * LSTR + col] = (__bf16)rv;
            }
        }
    }
    __syncthreads();

    // Phase B: proj GEMM + exact GELU + residual
#pragma unroll 1
    for (int pass = 0; pass < 2; ++pass) {
        v8f acc[8];
#pragma unroll
        for (int t = 0; t < 8; ++t) acc[t] = (v8f){0.f,0.f,0.f,0.f,0.f,0.f,0.f,0.f};
#pragma unroll
        for (int kk = 0; kk < 8; ++kk) {
            v16bf a = load_afrag_lds(retlds, 16 * w, lane, kk * 32);
#pragma unroll
            for (int t = 0; t < 8; ++t) {
                v16bf bb = load_bfrag(wproj_pack, (pass * 8 + t) * 8 + kk, lane);
                acc[t] = __builtin_amdgcn_wmma_f32_16x16x32_bf16(false, a, false, bb, (short)0, acc[t], false, false);
            }
        }
#pragma unroll
        for (int t = 0; t < 8; ++t) {
            const int col = (pass * 8 + t) * 16 + col0;
            const float bp = b_proj[col];
#pragma unroll
            for (int r = 0; r < 8; ++r) {
                const int lr = 16 * w + r + hi8;
                const size_t row = (size_t)row0 + lr;
                const float v0 = acc[t][r] + bp;
                const float g  = 0.5f * v0 * (1.0f + erff(v0 * 0.70710678118654752f));
                out[row * MIDC + col] = g + x[row * MIDC + col];
            }
        }
    }
}

// ---------------- host launcher ----------------
extern "C" void kernel_launch(void* const* d_in, const int* in_sizes, int n_in,
                              void* d_out, int out_size, void* d_ws, size_t ws_size,
                              hipStream_t stream) {
    (void)in_sizes; (void)n_in; (void)out_size; (void)ws_size;
    const float* x     = (const float*)d_in[0];
    const float* Wqkv  = (const float*)d_in[1];
    const float* bqkv  = (const float*)d_in[2];
    const float* klnw  = (const float*)d_in[3];
    const float* klnb  = (const float*)d_in[4];
    const float* vlnw  = (const float*)d_in[5];
    const float* vlnb  = (const float*)d_in[6];
    const float* Wproj = (const float*)d_in[7];
    const float* bproj = (const float*)d_in[8];
    float* out = (float*)d_out;

    char* ws = (char*)d_ws;
    __bf16* wkv_pack   = (__bf16*)(ws + WS_WKV);
    __bf16* wproj_pack = (__bf16*)(ws + WS_WPROJ);
    float*  kv_acc     = (float*)(ws + WS_KVACC);
    __bf16* weff_pack  = (__bf16*)(ws + WS_WEFF);
    float*  beff       = (float*)(ws + WS_BEFF);

    hipMemsetAsync(kv_acc, 0, (size_t)NBATCH * 4096 * sizeof(float), stream);

    pack_weights_kernel<<<48, 256, 0, stream>>>(Wqkv, Wproj, wkv_pack, wproj_pack);

    const size_t smem1 = 67584 + 16384 + 8 * 544 * sizeof(float);   // 101376
    qkv_ln_kv_kernel<<<NROWS / MBLK, 256, smem1, stream>>>(
        x, wkv_pack, bqkv, klnw, klnb, vlnw, vlnb, kv_acc);

    weff_kernel<<<128, 256, 0, stream>>>(Wqkv, kv_acc, weff_pack);
    beff_kernel<<<NBATCH, 256, 0, stream>>>(bqkv, kv_acc, beff);

    const size_t smem3 = 2 * 67584;                                  // 135168
    fused_out_kernel<<<NROWS / MBLK, 256, smem3, stream>>>(
        x, weff_pack, beff, wproj_pack, bproj, out);
}
